// MultiHeadAttention_14001593385609
// MI455X (gfx1250) — matrix operand order repaired and checked by reading, not compiled, not run on hardware
//
#include <hip/hip_runtime.h>
#include <hip/hip_bf16.h>
#include <math.h>

typedef __attribute__((ext_vector_type(16))) _Float16 v16h;
typedef __attribute__((ext_vector_type(8)))  float    v8f;

#define HID   768
#define NH    8
#define NKV   2
#define HD    96
#define SEQ   4096
#define BATCH 2
#define NQK   (HID + 2 * NKV * HD)   // 1152 = 768 q + 192 k + 192 v
#define MTOT  (BATCH * SEQ)          // 8192
#define QK_SCALE 0.1020620726159658f // 1/sqrt(96)

// ---------------------------------------------------------------------------
// CDNA5 async global->LDS copies (ASYNCcnt-tracked; no VGPR data round-trip).
// VDST holds the LDS byte offset; the low 32 bits of a generic __shared__
// pointer are exactly that offset (flat LDS aperture keeps offset in addr[31:0]).
// ---------------------------------------------------------------------------
__device__ inline void async_copy_b64(const void* gptr, void* lds) {
  unsigned loff = (unsigned)(size_t)lds;
  asm volatile("global_load_async_to_lds_b64 %0, %1, off"
               :: "v"(loff), "v"(gptr) : "memory");
}
__device__ inline void async_copy_b128(const void* gptr, void* lds) {
  unsigned loff = (unsigned)(size_t)lds;
  asm volatile("global_load_async_to_lds_b128 %0, %1, off"
               :: "v"(loff), "v"(gptr) : "memory");
}
__device__ inline void wait_async0() {
  asm volatile("s_wait_asynccnt 0" ::: "memory");
}

// ---------------------------------------------------------------------------
// WMMA fragment gathers.  16-bit A (16xK): lane l holds row M=l%16; half i maps
// to k = (l/16)*16 + i  -> 32B contiguous per lane.
// 16-bit B (Kx16) stored transposed [n][k]: lane l holds col N=l%16; halves are
// k = (l/16)*16 + i  -> 32B contiguous per lane.
// ---------------------------------------------------------------------------
__device__ inline v16h load_a_frag(const _Float16* base, int stride, int row,
                                   int kofs, int lane) {
  union { v16h v; uint4 q[2]; } r;
  const int g = (lane >> 4) * 16;
  const char* p = (const char*)(base + (size_t)row * stride + kofs + g);
  r.q[0] = *(const uint4*)(p);
  r.q[1] = *(const uint4*)(p + 16);   // k = kofs + 8 + g .. +7 (halves)
  return r.v;
}

__device__ inline v16h load_b_frag(const _Float16* base, int stride, int row,
                                   int kofs, int lane) {
  union { v16h v; uint4 q[2]; } r;
  const int k0 = kofs + (lane >> 4) * 16;
  const char* p = (const char*)(base + (size_t)row * stride + k0);
  r.q[0] = *(const uint4*)(p);
  r.q[1] = *(const uint4*)(p + 16);
  return r.v;
}

// ---------------------------------------------------------------------------
// Kernel 1: QKV projection.  Y[8192 x 1152] (f16) = X (f32) @ [Wq|Wk|Wv] (f32)
// Block tile 128x64, 8 waves in 4x2, each wave a 32x32 tile (2x2 WMMA frags).
// Staging converts f32->f16 (must pass through VALU, so synchronous loads).
// ---------------------------------------------------------------------------
__global__ __launch_bounds__(256)
void qkv_gemm_kernel(const float* __restrict__ X, const float* __restrict__ wq,
                     const float* __restrict__ wk, const float* __restrict__ wv,
                     _Float16* __restrict__ Y) {
  __shared__ __align__(16) _Float16 As[128 * 40];  // [m][k], padded stride 40
  __shared__ __align__(16) _Float16 Bs[64 * 40];   // [n][k], padded stride 40
  const int t    = threadIdx.x;
  const int lane = t & 31;
  const int wave = t >> 5;
  const int wm   = wave & 3;
  const int wn   = wave >> 2;
  const int row0 = blockIdx.x * 128;
  const int col0 = blockIdx.y * 64;

  v8f acc[2][2];
#pragma unroll
  for (int i = 0; i < 2; ++i)
#pragma unroll
    for (int j = 0; j < 2; ++j)
#pragma unroll
      for (int r = 0; r < 8; ++r) acc[i][j][r] = 0.0f;

  for (int k0 = 0; k0 < HID; k0 += 32) {
    // Stage A tile 128x32, converting f32 -> f16.
#pragma unroll
    for (int p = 0; p < 4; ++p) {
      const int row = p * 32 + (t >> 3);
      const int kq  = (t & 7) * 4;
      const float4 f =
          *(const float4*)(X + (size_t)(row0 + row) * HID + k0 + kq);
      _Float16* dst = As + row * 40 + kq;
      dst[0] = (_Float16)f.x; dst[1] = (_Float16)f.y;
      dst[2] = (_Float16)f.z; dst[3] = (_Float16)f.w;
    }
    // Stage B tile transposed: Bs[n][k] = W[k0+k][col0+n]  (f32 -> f16).
#pragma unroll
    for (int p = 0; p < 8; ++p) {
      const int k  = p * 4 + (t >> 6);
      const int n  = t & 63;
      const int gn = col0 + n;
      float f;
      if (gn < HID)            f = wq[(size_t)(k0 + k) * HID + gn];
      else if (gn < HID + 192) f = wk[(size_t)(k0 + k) * 192 + (gn - HID)];
      else                     f = wv[(size_t)(k0 + k) * 192 + (gn - HID - 192)];
      Bs[n * 40 + k] = (_Float16)f;
    }
    __syncthreads();

    v16h af[2], bf[2];
    af[0] = load_a_frag(As, 40, wm * 32 + (lane & 15), 0, lane);
    af[1] = load_a_frag(As, 40, wm * 32 + 16 + (lane & 15), 0, lane);
    bf[0] = load_b_frag(Bs, 40, wn * 32 + (lane & 15), 0, lane);
    bf[1] = load_b_frag(Bs, 40, wn * 32 + 16 + (lane & 15), 0, lane);
#pragma unroll
    for (int i = 0; i < 2; ++i)
#pragma unroll
      for (int j = 0; j < 2; ++j)
        acc[i][j] = __builtin_amdgcn_wmma_f32_16x16x32_f16(
            false, af[i], false, bf[j], (short)0, acc[i][j], false, false);
    __syncthreads();
  }

  // Epilogue: C frag -> f16 Y.  VGPR r, lanes 0-15: M=r; lanes 16-31: M=8+r.
#pragma unroll
  for (int i = 0; i < 2; ++i)
#pragma unroll
    for (int j = 0; j < 2; ++j) {
      const int m = row0 + wm * 32 + i * 16 + ((lane >> 4) << 3);
      const int n = col0 + wn * 32 + j * 16 + (lane & 15);
#pragma unroll
      for (int r = 0; r < 8; ++r)
        Y[(size_t)(m + r) * NQK + n] = (_Float16)acc[i][j][r];
    }
}

// ---------------------------------------------------------------------------
// Kernel 2: RoPE + pack.  Reads Y [8192 x 1152] f16, applies full-dim rotary
// to Q/K (fold 1/sqrt(96) into Q), writes:
//   Qh [B,NH, S, 96]   row-major (A-side of Q.K^T)
//   Kh [B,NKV,S, 96]   row-major (B-side of Q.K^T: Bt[n=key][k=d])
//   Vt [B,NKV,96, S]   V transposed (B-side of P.V: Bt[n=d][k=key])
// ---------------------------------------------------------------------------
__global__ __launch_bounds__(256)
void rope_pack_kernel(const _Float16* __restrict__ Y, _Float16* __restrict__ Qh,
                      _Float16* __restrict__ Kh, _Float16* __restrict__ Vt) {
  const size_t idx = (size_t)blockIdx.x * 256 + threadIdx.x;
  if (idx >= (size_t)MTOT * NQK) return;
  const int m   = (int)(idx / NQK);
  const int col = (int)(idx - (size_t)m * NQK);
  const int b   = m / SEQ;
  const int s   = m - b * SEQ;
  const float v = (float)Y[idx];

  if (col < HID) {                      // ---- Q, with RoPE + score scale
    const int h = col / HD, d = col - h * HD;
    const int jj = (d < 48) ? d : d - 48;
    const float ang = (float)s * __powf(10000.0f, -(2.0f * jj) / 96.0f);
    float sn, cs; __sincosf(ang, &sn, &cs);
    const float rot = (d < 48) ? -(float)Y[idx + 48] : (float)Y[idx - 48];
    const float out = (v * cs + rot * sn) * QK_SCALE;
    Qh[((size_t)(b * NH + h) * SEQ + s) * HD + d] = (_Float16)out;
  } else if (col < HID + NKV * HD) {    // ---- K, with RoPE
    const int c2 = col - HID;
    const int hk = c2 / HD, d = c2 - hk * HD;
    const int jj = (d < 48) ? d : d - 48;
    const float ang = (float)s * __powf(10000.0f, -(2.0f * jj) / 96.0f);
    float sn, cs; __sincosf(ang, &sn, &cs);
    const float rot = (d < 48) ? -(float)Y[idx + 48] : (float)Y[idx - 48];
    Kh[((size_t)(b * NKV + hk) * SEQ + s) * HD + d] = (_Float16)(v * cs + rot * sn);
  } else {                              // ---- V, transposed store
    const int c2 = col - HID - NKV * HD;
    const int hk = c2 / HD, d = c2 - hk * HD;
    Vt[((size_t)(b * NKV + hk) * HD + d) * SEQ + s] = (_Float16)v;
  }
}

// ---------------------------------------------------------------------------
// Kernel 3: flash attention.  grid = (S/128, B*NH); 8 waves x 16 queries.
// Key loop: 64 keys/tile, staged via CDNA5 async global->LDS (ASYNCcnt).
// Online softmax in f32 using the C fragment layout (row = VGPR idx +
// 8*(lane/16); shfl_xor reductions across the 16-lane half).  P round-trips
// through wave-private LDS to re-layout C-frag -> A-frag (LDS in-order/wave).
// 24 WMMA per wave per 64-key tile.
// ---------------------------------------------------------------------------
__global__ __launch_bounds__(256)
void flash_attn_kernel(const _Float16* __restrict__ Qh,
                       const _Float16* __restrict__ Kh,
                       const _Float16* __restrict__ Vt,
                       _Float16* __restrict__ Oa) {
  __shared__ __align__(16) _Float16 Ks[64 * 104];      // [key][d],  pad 104
  __shared__ __align__(16) _Float16 Vs[96 * 72];       // [d][key],  pad 72
  __shared__ __align__(16) _Float16 Ps[8 * 16 * 72];   // per-wave P, pad 72
  const int t = threadIdx.x, lane = t & 31, wave = t >> 5;
  const int bh = blockIdx.y, b = bh >> 3, h = bh & 7, hk = h >> 2;
  const int q0 = blockIdx.x * 128 + wave * 16;

  const _Float16* Qp = Qh + ((size_t)(b * NH + h) * SEQ + q0) * HD;
  const _Float16* Kp = Kh + (size_t)(b * NKV + hk) * SEQ * HD;
  const _Float16* Vp = Vt + (size_t)(b * NKV + hk) * HD * SEQ;

  v16h qa[3];
#pragma unroll
  for (int kk = 0; kk < 3; ++kk)
    qa[kk] = load_a_frag(Qp, HD, lane & 15, kk * 32, lane);

  v8f oacc[6];
  float mrow[8], lrow[8];
#pragma unroll
  for (int j = 0; j < 6; ++j)
#pragma unroll
    for (int r = 0; r < 8; ++r) oacc[j][r] = 0.0f;
#pragma unroll
  for (int r = 0; r < 8; ++r) { mrow[r] = -1e30f; lrow[r] = 0.0f; }

  _Float16* Pw = Ps + wave * 16 * 72;

  for (int kt = 0; kt < SEQ; kt += 64) {
    // Async-stage K (64x96) and Vt (96x64): 6x b64 per thread per tensor.
#pragma unroll
    for (int p = 0; p < 6; ++p) {
      const int off = (p * 256 + t) * 4;          // 0..6140, step 4
      const int key = off / 96, d = off - key * 96;
      async_copy_b64(Kp + (size_t)(kt + key) * HD + d, Ks + key * 104 + d);
      const int dv = off >> 6, sv = off & 63;
      async_copy_b64(Vp + (size_t)dv * SEQ + kt + sv, Vs + dv * 72 + sv);
    }
    wait_async0();
    __syncthreads();

    // Scores: S[16 x 64] = Q(16x96) . K^T  (4 n-frags x 3 k-steps)
    v8f sc2[4];
#pragma unroll
    for (int f = 0; f < 4; ++f) {
#pragma unroll
      for (int r = 0; r < 8; ++r) sc2[f][r] = 0.0f;
#pragma unroll
      for (int kk = 0; kk < 3; ++kk) {
        v16h bk = load_b_frag(Ks, 104, f * 16 + (lane & 15), kk * 32, lane);
        sc2[f] = __builtin_amdgcn_wmma_f32_16x16x32_f16(
            false, qa[kk], false, bk, (short)0, sc2[f], false, false);
      }
    }

    // Online softmax + write P (f16) into wave-private LDS.
#pragma unroll
    for (int r = 0; r < 8; ++r) {
      float tv0 = sc2[0][r], tv1 = sc2[1][r], tv2 = sc2[2][r], tv3 = sc2[3][r];
      float mt = fmaxf(fmaxf(tv0, tv1), fmaxf(tv2, tv3));
#pragma unroll
      for (int msk = 8; msk >= 1; msk >>= 1)
        mt = fmaxf(mt, __shfl_xor(mt, msk, 32));
      const float mnew = fmaxf(mrow[r], mt);
      const float corr = __expf(mrow[r] - mnew);
      const float p0 = __expf(tv0 - mnew), p1 = __expf(tv1 - mnew);
      const float p2 = __expf(tv2 - mnew), p3 = __expf(tv3 - mnew);
      float rs = (p0 + p1) + (p2 + p3);
#pragma unroll
      for (int msk = 8; msk >= 1; msk >>= 1) rs += __shfl_xor(rs, msk, 32);
      lrow[r] = lrow[r] * corr + rs;
      mrow[r] = mnew;
#pragma unroll
      for (int j = 0; j < 6; ++j) oacc[j][r] *= corr;
      const int m = r + ((lane >> 4) << 3);
      _Float16* pr = Pw + m * 72 + (lane & 15);
      pr[0]  = (_Float16)p0;
      pr[16] = (_Float16)p1;
      pr[32] = (_Float16)p2;
      pr[48] = (_Float16)p3;
    }

    // O += P(16x64) . V(64x96): 2 k-frags x 6 n-frags over d.
    v16h pa0 = load_a_frag(Pw, 72, lane & 15, 0, lane);
    v16h pa1 = load_a_frag(Pw, 72, lane & 15, 32, lane);
#pragma unroll
    for (int j = 0; j < 6; ++j) {
      v16h bv0 = load_b_frag(Vs, 72, j * 16 + (lane & 15), 0, lane);
      v16h bv1 = load_b_frag(Vs, 72, j * 16 + (lane & 15), 32, lane);
      oacc[j] = __builtin_amdgcn_wmma_f32_16x16x32_f16(
          false, pa0, false, bv0, (short)0, oacc[j], false, false);
      oacc[j] = __builtin_amdgcn_wmma_f32_16x16x32_f16(
          false, pa1, false, bv1, (short)0, oacc[j], false, false);
    }
    __syncthreads();
  }

  // Normalize and store to attn buffer [B*S][768] f16, col = h*96 + d.
#pragma unroll
  for (int j = 0; j < 6; ++j) {
    const int n = h * HD + j * 16 + (lane & 15);
#pragma unroll
    for (int r = 0; r < 8; ++r) {
      const int srow = q0 + r + ((lane >> 4) << 3);
      Oa[((size_t)b * SEQ + srow) * HID + n] = (_Float16)(oacc[j][r] / lrow[r]);
    }
  }
}

// ---------------------------------------------------------------------------
// Kernel 4: output projection.  out[8192 x 768] f32 = attn (f16) @ Wo (f32).
// A tile is a pure f16 copy -> async global->LDS b128; B converts via VALU.
// ---------------------------------------------------------------------------
__global__ __launch_bounds__(256)
void out_gemm_kernel(const _Float16* __restrict__ A, const float* __restrict__ wo,
                     float* __restrict__ out) {
  __shared__ __align__(16) _Float16 As[128 * 40];
  __shared__ __align__(16) _Float16 Bs[64 * 40];
  const int t    = threadIdx.x;
  const int lane = t & 31;
  const int wave = t >> 5;
  const int wm   = wave & 3;
  const int wn   = wave >> 2;
  const int row0 = blockIdx.x * 128;
  const int col0 = blockIdx.y * 64;

  v8f acc[2][2];
#pragma unroll
  for (int i = 0; i < 2; ++i)
#pragma unroll
    for (int j = 0; j < 2; ++j)
#pragma unroll
      for (int r = 0; r < 8; ++r) acc[i][j][r] = 0.0f;

  for (int k0 = 0; k0 < HID; k0 += 32) {
    // Async-stage A tile 128x32 halves: 2x b128 per thread.
#pragma unroll
    for (int p = 0; p < 2; ++p) {
      const int row = p * 64 + (t >> 2);
      const int kq  = (t & 3) * 8;
      async_copy_b128(A + (size_t)(row0 + row) * HID + k0 + kq,
                      As + row * 40 + kq);
    }
    // Stage B tile transposed, f32 -> f16.
#pragma unroll
    for (int p = 0; p < 8; ++p) {
      const int k = p * 4 + (t >> 6);
      const int n = t & 63;
      Bs[n * 40 + k] = (_Float16)wo[(size_t)(k0 + k) * HID + col0 + n];
    }
    wait_async0();
    __syncthreads();

    v16h af[2], bf[2];
    af[0] = load_a_frag(As, 40, wm * 32 + (lane & 15), 0, lane);
    af[1] = load_a_frag(As, 40, wm * 32 + 16 + (lane & 15), 0, lane);
    bf[0] = load_b_frag(Bs, 40, wn * 32 + (lane & 15), 0, lane);
    bf[1] = load_b_frag(Bs, 40, wn * 32 + 16 + (lane & 15), 0, lane);
#pragma unroll
    for (int i = 0; i < 2; ++i)
#pragma unroll
      for (int j = 0; j < 2; ++j)
        acc[i][j] = __builtin_amdgcn_wmma_f32_16x16x32_f16(
            false, af[i], false, bf[j], (short)0, acc[i][j], false, false);
    __syncthreads();
  }

#pragma unroll
  for (int i = 0; i < 2; ++i)
#pragma unroll
    for (int j = 0; j < 2; ++j) {
      const int m = row0 + wm * 32 + i * 16 + ((lane >> 4) << 3);
      const int n = col0 + wn * 32 + j * 16 + (lane & 15);
#pragma unroll
      for (int r = 0; r < 8; ++r)
        out[(size_t)(m + r) * HID + n] = acc[i][j][r];
    }
}

// ---------------------------------------------------------------------------
extern "C" void kernel_launch(void* const* d_in, const int* in_sizes, int n_in,
                              void* d_out, int out_size, void* d_ws, size_t ws_size,
                              hipStream_t stream) {
  const float* X  = (const float*)d_in[0];
  const float* wq = (const float*)d_in[1];
  const float* wk = (const float*)d_in[2];
  const float* wv = (const float*)d_in[3];
  const float* wo = (const float*)d_in[4];
  float* out = (float*)d_out;

  // Workspace layout (f16):
  //   Yt  : MTOT x 1152  (QKV gemm result; later reused as attn buffer)
  //   Qh  : MTOT x 768
  //   Kh  : MTOT x 192
  //   Vt  : MTOT x 192
  _Float16* Yt = (_Float16*)d_ws;
  _Float16* Qh = Yt + (size_t)MTOT * NQK;
  _Float16* Kh = Qh + (size_t)MTOT * HID;
  _Float16* Vt = Kh + (size_t)MTOT * (NKV * HD);
  _Float16* Ah = Yt;  // attn output aliases Yt (Yt dead after rope_pack)

  // 1) QKV projection
  qkv_gemm_kernel<<<dim3(MTOT / 128, NQK / 64), 256, 0, stream>>>(X, wq, wk, wv, Yt);
  // 2) RoPE + pack (also scales Q by 1/sqrt(96))
  {
    const size_t tot = (size_t)MTOT * NQK;
    rope_pack_kernel<<<dim3((unsigned)((tot + 255) / 256)), 256, 0, stream>>>(Yt, Qh, Kh, Vt);
  }
  // 3) flash attention
  flash_attn_kernel<<<dim3(SEQ / 128, BATCH * NH), 256, 0, stream>>>(Qh, Kh, Vt, Ah);
  // 4) output projection
  out_gemm_kernel<<<dim3(MTOT / 128, HID / 64), 256, 0, stream>>>(Ah, wo, out);
}
